// MultiHeadAttention_68556267979359
// MI455X (gfx1250) — matrix operand order repaired and checked by reading, not compiled, not run on hardware
//
#include <hip/hip_runtime.h>
#include <hip/hip_bf16.h>

typedef __attribute__((ext_vector_type(16))) __bf16 v16bf;
typedef __attribute__((ext_vector_type(8)))  __bf16 v8bf;
typedef __attribute__((ext_vector_type(4)))  __bf16 v4bf;
typedef __attribute__((ext_vector_type(8)))  float  v8f;
typedef __attribute__((ext_vector_type(4)))  float  v4f;
typedef __attribute__((ext_vector_type(4)))  int    v4i;

#if __has_builtin(__builtin_amdgcn_global_load_async_to_lds_b128)
#define USE_ASYNC_LDS 1
#endif

typedef __attribute__((address_space(1))) v4i* gv4i_p;
typedef __attribute__((address_space(3))) v4i* lv4i_p;

static __device__ __forceinline__ void ld_async_b128(const void* g, void* s) {
#ifdef USE_ASYNC_LDS
  __builtin_amdgcn_global_load_async_to_lds_b128((gv4i_p)g, (lv4i_p)s, 0, 0);
#else
  *(v8bf*)s = *(const v8bf*)g;
#endif
}
static __device__ __forceinline__ void wait_async() {
#ifdef USE_ASYNC_LDS
#if __has_builtin(__builtin_amdgcn_s_wait_asynccnt)
  __builtin_amdgcn_s_wait_asynccnt(0);
#else
  asm volatile("s_wait_asynccnt 0" ::: "memory");
#endif
#endif
}

static __device__ __forceinline__ v8f wmma_bf16(v16bf a, v16bf b, v8f c) {
  return __builtin_amdgcn_wmma_f32_16x16x32_bf16(false, a, false, b, (short)0, c, false, false);
}

static __device__ __forceinline__ v16bf cat8(v8bf lo, v8bf hi) {
  v16bf r;
#pragma unroll
  for (int i = 0; i < 8; ++i) { r[i] = lo[i]; r[i + 8] = hi[i]; }
  return r;
}

// A-matrix 16x32 bf16 fragment. base points at A[row0][kchunk0], ld in elements.
// lane (m = l&15, g = l>>4): elems 0..7 = A[m][g*8+0..7], 8..15 = A[m][16+g*8+0..7]
static __device__ __forceinline__ v16bf afrag_bf(const __bf16* base, int ld, int m, int g) {
  const __bf16* p = base + (size_t)m * ld + g * 8;
  return cat8(*(const v8bf*)p, *(const v8bf*)(p + 16));
}
// B-matrix 32x16 fragment with B[k][n] = W[n][k] (W row-major [N,K]):
// lane (n = l&15, g = l>>4): elems 0..7 = W[n][g*8+0..7], 8..15 = W[n][16+g*8+0..7]
static __device__ __forceinline__ v16bf bfrag_bf(const __bf16* base, int ld, int n, int g) {
  const __bf16* p = base + (size_t)n * ld + g * 8;
  return cat8(*(const v8bf*)p, *(const v8bf*)(p + 16));
}

// ---------------- Kernel 0: f32 -> bf16 stream conversion --------------------
__global__ __launch_bounds__(256) void cvt_kernel(
    const float* __restrict__ src, __bf16* __restrict__ dst, int n) {
  int i = (blockIdx.x * 256 + threadIdx.x) * 8;
  if (i >= n) return;
  v4f a = *(const v4f*)(src + i);
  v4f b = *(const v4f*)(src + i + 4);
  v8bf r;
#pragma unroll
  for (int j = 0; j < 4; ++j) { r[j] = (__bf16)a[j]; r[j + 4] = (__bf16)b[j]; }
  *(v8bf*)(dst + i) = r;
}

// ---------------- Kernel 1: projections --------------------------------------
// out[row, col] = sum_d src[row,d] * W[col,d];  rows = B*T = 16384, cols = 512
// Each wave: 16 rows x 64 cols, software-pipelined over the K chunks.
__global__ __launch_bounds__(256) void proj_kernel(
    const __bf16* __restrict__ src, const __bf16* __restrict__ W,
    __bf16* __restrict__ dst, float scale, int transpose_out) {
  const int lane = threadIdx.x & 31, wave = threadIdx.x >> 5;
  const int l15 = lane & 15, g = lane >> 4;
  int tile = blockIdx.x * 8 + wave;          // 8192 macro-tiles
  int row0 = (tile >> 3) * 16;               // 1024 row tiles
  int cg0 = (tile & 7) * 64;                 // 8 col groups of 64
  const __bf16* abase = src + (size_t)row0 * 512;
  const __bf16* wbase = W + (size_t)cg0 * 512;
  v8f acc[4] = {};
  v16bf a_c = afrag_bf(abase, 512, l15, g);
  v16bf b_c[4];
#pragma unroll
  for (int j = 0; j < 4; ++j) b_c[j] = bfrag_bf(wbase + (size_t)j * 16 * 512, 512, l15, g);
#pragma unroll
  for (int kc = 0; kc < 16; ++kc) {
    v16bf a_n = a_c;
    v16bf b_n[4] = {b_c[0], b_c[1], b_c[2], b_c[3]};
    if (kc < 15) {
      a_n = afrag_bf(abase + (kc + 1) * 32, 512, l15, g);
#pragma unroll
      for (int j = 0; j < 4; ++j)
        b_n[j] = bfrag_bf(wbase + (size_t)j * 16 * 512 + (kc + 1) * 32, 512, l15, g);
    }
#pragma unroll
    for (int j = 0; j < 4; ++j) acc[j] = wmma_bf16(a_c, b_c[j], acc[j]);
    a_c = a_n;
#pragma unroll
    for (int j = 0; j < 4; ++j) b_c[j] = b_n[j];
  }
  int b = row0 >> 9, t0 = row0 & 511;
#pragma unroll
  for (int j = 0; j < 4; ++j) {
    int col = cg0 + j * 16 + l15, h = col >> 6;
    if (!transpose_out) {
      size_t base = ((size_t)b * 8 + h) * 32768 + (size_t)(col & 63);
#pragma unroll
      for (int r = 0; r < 8; ++r)
        dst[base + (size_t)(t0 + r + 8 * g) * 64] = (__bf16)(acc[j][r] * scale);
    } else {
      v8bf pk;
#pragma unroll
      for (int r = 0; r < 8; ++r) pk[r] = (__bf16)(acc[j][r] * scale);
      size_t idx = (((size_t)b * 8 + h) * 64 + (col & 63)) * 512 + t0 + 8 * g;
      *(v8bf*)(dst + idx) = pk;
    }
  }
}

// ---------------- Kernel 2: attention ----------------------------------------
// Workgroup (4 waves): stages the (b,h) K slice into LDS via async loads.
// Per wave: 16 q rows. S = Q K^T -> LDS(f32), softmax -> LDS(bf16 P),
// O = P V, store O^T into Xs (implements the reference's layout scramble).
__global__ __launch_bounds__(128) void attn_kernel(
    const __bf16* __restrict__ Qb,  // [BH,512,64] pre-scaled by 1/8
    const __bf16* __restrict__ Kb,  // [BH,512,64]
    const __bf16* __restrict__ Vt,  // [BH,64,512]
    __bf16* __restrict__ Xs) {      // [B,512,512] = O^T per batch
  __shared__ float  sc[4 * 16 * 512];   // 128 KB scores
  __shared__ __bf16 pb[4 * 16 * 512];   //  64 KB probabilities
  __shared__ __bf16 kb_s[512 * 64];     //  64 KB staged K slice
  __shared__ float  rinv_s[4][16];
  const int lane = threadIdx.x & 31, w = threadIdx.x >> 5;
  const int l15 = lane & 15, g = lane >> 4;
  int bh = blockIdx.x >> 3;
  int tq0 = (blockIdx.x & 7) * 64 + w * 16;
  const __bf16* Q = Qb + (size_t)bh * 512 * 64;
  const __bf16* K = Kb + (size_t)bh * 512 * 64;
  const __bf16* V = Vt + (size_t)bh * 64 * 512;
  float*  S = sc + w * 16 * 512;
  __bf16* P = pb + w * 16 * 512;

  // stage K (64 KB) into LDS with async copies: 128 threads x 32 x 16B
#pragma unroll 4
  for (int i = 0; i < 32; ++i) {
    int off = ((int)threadIdx.x + i * 128) * 8;   // elements (16B each)
    ld_async_b128(K + off, kb_s + off);
  }
  wait_async();
  __syncthreads();

  v16bf aq0 = afrag_bf(Q + (size_t)tq0 * 64, 64, l15, g);
  v16bf aq1 = afrag_bf(Q + (size_t)tq0 * 64 + 32, 64, l15, g);

  for (int tk = 0; tk < 32; ++tk) {
    v8f acc = {};
    acc = wmma_bf16(aq0, bfrag_bf(kb_s + tk * 1024, 64, l15, g), acc);
    acc = wmma_bf16(aq1, bfrag_bf(kb_s + tk * 1024 + 32, 64, l15, g), acc);
#pragma unroll
    for (int r = 0; r < 8; ++r)
      S[(r + 8 * g) * 512 + tk * 16 + l15] = acc[r];
  }
  // intra-wave softmax: lane (l15,g) handles row l15, columns g*256..g*256+255
  {
    const v4f* s4 = (const v4f*)(S + l15 * 512 + g * 256);
    v4bf* p4 = (v4bf*)(P + l15 * 512 + g * 256);
    float mx = -1e30f;
#pragma unroll 4
    for (int i = 0; i < 64; ++i) {
      v4f x = s4[i];
      mx = fmaxf(mx, fmaxf(fmaxf(x[0], x[1]), fmaxf(x[2], x[3])));
    }
    mx = fmaxf(mx, __shfl_xor(mx, 16));
    float sum = 0.f;
#pragma unroll 4
    for (int i = 0; i < 64; ++i) {
      v4f x = s4[i];
      v4bf e4;
#pragma unroll
      for (int j = 0; j < 4; ++j) {
        float e = __expf(x[j] - mx);
        sum += e;
        e4[j] = (__bf16)e;
      }
      p4[i] = e4;
    }
    sum += __shfl_xor(sum, 16);
    if (g == 0) rinv_s[w][l15] = 1.0f / sum;
  }
  float rinvv[8];
#pragma unroll
  for (int r = 0; r < 8; ++r) rinvv[r] = rinv_s[w][r + 8 * g];

  // O = P V : A fragment (LDS) shared by the 4 dv tiles; pipelined V loads
  v8f acc[4] = {};
  v16bf a_c = afrag_bf(P, 512, l15, g);
  v16bf b_c[4];
#pragma unroll
  for (int dt = 0; dt < 4; ++dt) b_c[dt] = bfrag_bf(V + dt * 16 * 512, 512, l15, g);
#pragma unroll 4
  for (int kc = 0; kc < 16; ++kc) {
    v16bf a_n = a_c;
    v16bf b_n[4] = {b_c[0], b_c[1], b_c[2], b_c[3]};
    if (kc < 15) {
      a_n = afrag_bf(P + (kc + 1) * 32, 512, l15, g);
#pragma unroll
      for (int dt = 0; dt < 4; ++dt)
        b_n[dt] = bfrag_bf(V + dt * 16 * 512 + (kc + 1) * 32, 512, l15, g);
    }
#pragma unroll
    for (int dt = 0; dt < 4; ++dt) acc[dt] = wmma_bf16(a_c, b_c[dt], acc[dt]);
    a_c = a_n;
#pragma unroll
    for (int dt = 0; dt < 4; ++dt) b_c[dt] = b_n[dt];
  }
  int bb = bh >> 3, h = bh & 7;
#pragma unroll
  for (int dt = 0; dt < 4; ++dt) {
    int e = h * 64 + dt * 16 + l15;     // output feature (row of Xs)
    v8bf pk;
#pragma unroll
    for (int r = 0; r < 8; ++r) pk[r] = (__bf16)(acc[dt][r] * rinvv[r]);
    *(v8bf*)(Xs + ((size_t)bb * 512 + e) * 512 + tq0 + 8 * g) = pk;
  }
}

// ---------------- Kernel 3: fc + residual + layernorm ------------------------
__global__ __launch_bounds__(256) void fc_ln_kernel(
    const __bf16* __restrict__ Xs, const __bf16* __restrict__ Wfc,
    const float* __restrict__ res, const float* __restrict__ gamma,
    const float* __restrict__ beta, float* __restrict__ out) {
  __shared__ float xb[16 * 512];           // 32 KB
  const int lane = threadIdx.x & 31, w = threadIdx.x >> 5;
  const int l15 = lane & 15, g = lane >> 4;
  int b = blockIdx.x >> 5, t0 = (blockIdx.x & 31) * 16;
  const __bf16* A = Xs + ((size_t)b * 512 + t0) * 512;
  const __bf16* wbase = Wfc + (size_t)w * 64 * 512;

  v8f acc[4] = {};
  v16bf a_c = afrag_bf(A, 512, l15, g);
  v16bf b_c[4];
#pragma unroll
  for (int nt = 0; nt < 4; ++nt) b_c[nt] = bfrag_bf(wbase + (size_t)nt * 16 * 512, 512, l15, g);
#pragma unroll
  for (int kc = 0; kc < 16; ++kc) {
    v16bf a_n = a_c;
    v16bf b_n[4] = {b_c[0], b_c[1], b_c[2], b_c[3]};
    if (kc < 15) {
      a_n = afrag_bf(A + (kc + 1) * 32, 512, l15, g);
#pragma unroll
      for (int nt = 0; nt < 4; ++nt)
        b_n[nt] = bfrag_bf(wbase + (size_t)nt * 16 * 512 + (kc + 1) * 32, 512, l15, g);
    }
#pragma unroll
    for (int nt = 0; nt < 4; ++nt) acc[nt] = wmma_bf16(a_c, b_c[nt], acc[nt]);
    a_c = a_n;
#pragma unroll
    for (int nt = 0; nt < 4; ++nt) b_c[nt] = b_n[nt];
  }
#pragma unroll
  for (int nt = 0; nt < 4; ++nt) {
    int col = w * 64 + nt * 16 + l15;
#pragma unroll
    for (int r = 0; r < 8; ++r) {
      int rr = r + 8 * g;
      xb[rr * 512 + col] = acc[nt][r] + res[((size_t)b * 512 + t0 + rr) * 512 + col];
    }
  }
  __syncthreads();
  // layernorm: wave w owns rows 2w, 2w+1; 16 columns per lane
  for (int rr = 2 * w; rr < 2 * w + 2; ++rr) {
    const float* x = xb + rr * 512;
    float s = 0.f, sq = 0.f;
#pragma unroll 4
    for (int i = lane * 16; i < lane * 16 + 16; ++i) { float v = x[i]; s += v; sq += v * v; }
#pragma unroll
    for (int off = 16; off > 0; off >>= 1) { s += __shfl_xor(s, off); sq += __shfl_xor(sq, off); }
    float mu = s * (1.0f / 512.0f);
    float rstd = rsqrtf(sq * (1.0f / 512.0f) - mu * mu + 1e-6f);
    size_t ob = ((size_t)b * 512 + t0 + rr) * 512;
#pragma unroll 4
    for (int i = lane * 16; i < lane * 16 + 16; ++i)
      out[ob + i] = (x[i] - mu) * rstd * gamma[i] + beta[i];
  }
}

extern "C" void kernel_launch(void* const* d_in, const int* in_sizes, int n_in,
                              void* d_out, int out_size, void* d_ws, size_t ws_size,
                              hipStream_t stream) {
  (void)in_sizes; (void)n_in; (void)out_size; (void)ws_size;
  const float* q     = (const float*)d_in[0];
  const float* k     = (const float*)d_in[1];
  const float* v     = (const float*)d_in[2];
  const float* w_q   = (const float*)d_in[3];
  const float* w_k   = (const float*)d_in[4];
  const float* w_v   = (const float*)d_in[5];
  const float* w_fc  = (const float*)d_in[6];
  const float* gamma = (const float*)d_in[7];
  const float* beta  = (const float*)d_in[8];

  const size_t BIG = (size_t)32 * 512 * 512;   // 8 Mi elems
  const size_t WSZ = (size_t)512 * 512;        // 256 Ki elems
  __bf16* p = (__bf16*)d_ws;
  __bf16 *Qb = p, *Kb = p + BIG, *Vt = p + 2 * BIG, *Xs = p + 3 * BIG;
  __bf16 *xq = p + 4 * BIG, *xk = p + 5 * BIG, *xv = p + 6 * BIG;
  __bf16 *wqB = p + 7 * BIG, *wkB = wqB + WSZ, *wvB = wkB + WSZ, *wfB = wvB + WSZ;

  const int nBig = (int)BIG, nW = (int)WSZ;
  cvt_kernel<<<nBig / (256 * 8), 256, 0, stream>>>(q, xq, nBig);
  cvt_kernel<<<nBig / (256 * 8), 256, 0, stream>>>(k, xk, nBig);
  cvt_kernel<<<nBig / (256 * 8), 256, 0, stream>>>(v, xv, nBig);
  cvt_kernel<<<nW / (256 * 8), 256, 0, stream>>>(w_q, wqB, nW);
  cvt_kernel<<<nW / (256 * 8), 256, 0, stream>>>(w_k, wkB, nW);
  cvt_kernel<<<nW / (256 * 8), 256, 0, stream>>>(w_v, wvB, nW);
  cvt_kernel<<<nW / (256 * 8), 256, 0, stream>>>(w_fc, wfB, nW);

  proj_kernel<<<1024, 256, 0, stream>>>(xq, wqB, Qb, 0.125f, 0);  // 1/sqrt(64)
  proj_kernel<<<1024, 256, 0, stream>>>(xk, wkB, Kb, 1.0f, 0);
  proj_kernel<<<1024, 256, 0, stream>>>(xv, wvB, Vt, 1.0f, 1);
  attn_kernel<<<2048, 128, 0, stream>>>(Qb, Kb, Vt, Xs);
  fc_ln_kernel<<<1024, 256, 0, stream>>>(Xs, wfB, q, gamma, beta, (float*)d_out);
}
